// MoELayer_20246475833795
// MI455X (gfx1250) — compile-verified, matrix operand order checked by reading, not run on hardware
//
#include <hip/hip_runtime.h>
#include <hip/hip_bf16.h>

// ---- fixed problem shape (from reference setup_inputs) ----
constexpr int T_ = 4096;   // tokens = B*S
constexpr int H_ = 1024;   // hidden
constexpr int F_ = 4096;   // ffn dim
constexpr int E_ = 8;      // experts
constexpr int R_ = 2 * T_; // total routed rows (top-2)

typedef __attribute__((ext_vector_type(16))) __bf16        v16bf;
typedef __attribute__((ext_vector_type(8)))  float         v8f;
typedef __attribute__((ext_vector_type(8)))  unsigned int  v8ui;

// fp32 -> bf16, round-to-nearest-even (scalar bit-trick, epilogue only)
__device__ __forceinline__ unsigned short f2bf(float f) {
    unsigned int u = __float_as_uint(f);
    u += 0x7FFFu + ((u >> 16) & 1u);
    return (unsigned short)(u >> 16);
}
// packed 2xfp32 -> 2xbf16 in one VALU op (CDNA5 packed-BF16 cvt, RNE).
// low 16 bits <- a, high 16 bits <- b.
__device__ __forceinline__ unsigned int pk2(float a, float b) {
    unsigned int r;
    asm("v_cvt_pk_bf16_f32 %0, %1, %2" : "=v"(r) : "v"(a), "v"(b));
    return r;
}

// ---------------- init: zero expert counters ----------------
__global__ void moe_init(int* counts, int* counts2) {
    if (threadIdx.x < E_) { counts[threadIdx.x] = 0; counts2[threadIdx.x] = 0; }
}

// ---------------- router: one wave32 per token ----------------
__global__ __launch_bounds__(256) void moe_router(
    const float* __restrict__ x, const float* __restrict__ Wr,
    const float* __restrict__ br, int* __restrict__ tope,
    float* __restrict__ topw, int* __restrict__ counts)
{
    const int t    = blockIdx.x * 8 + (threadIdx.x >> 5);
    const int lane = threadIdx.x & 31;
    float acc[E_];
#pragma unroll
    for (int e = 0; e < E_; ++e) acc[e] = 0.f;
    const float* xr = x + (size_t)t * H_;
    for (int j = 0; j < H_ / 32; ++j) {
        float xv = xr[j * 32 + lane];
        const float* w = Wr + (size_t)(j * 32 + lane) * E_;
#pragma unroll
        for (int e = 0; e < E_; ++e) acc[e] = fmaf(xv, w[e], acc[e]);
    }
#pragma unroll
    for (int off = 16; off > 0; off >>= 1)
#pragma unroll
        for (int e = 0; e < E_; ++e) acc[e] += __shfl_xor(acc[e], off, 32);

    if (lane == 0) {
        float lg[E_];
#pragma unroll
        for (int e = 0; e < E_; ++e) lg[e] = acc[e] + br[e];
        int i1 = 0;
#pragma unroll
        for (int e = 1; e < E_; ++e) if (lg[e] > lg[i1]) i1 = e;
        int i2 = (i1 == 0) ? 1 : 0;
#pragma unroll
        for (int e = 0; e < E_; ++e) if (e != i1 && lg[e] > lg[i2]) i2 = e;
        // renormalized top-2 weights: softmax normalizer cancels
        float w1 = 1.f / (1.f + expf(lg[i2] - lg[i1]));
        tope[t * 2 + 0] = i1;  topw[t * 2 + 0] = w1;
        tope[t * 2 + 1] = i2;  topw[t * 2 + 1] = 1.f - w1;
        atomicAdd(&counts[i1], 1);
        atomicAdd(&counts[i2], 1);
    }
}

// ---------------- exclusive scan of counts (E=8) ----------------
__global__ void moe_scan(const int* counts, int* base) {
    if (threadIdx.x == 0) {
        int s = 0;
        for (int e = 0; e < E_; ++e) { base[e] = s; s += counts[e]; }
    }
}

// ---------------- scatter tokens into per-expert compacted rows ----------------
__global__ void moe_scatter(const int* __restrict__ tope, const float* __restrict__ topw,
                            const int* __restrict__ base, int* __restrict__ counts2,
                            int* __restrict__ list, float* __restrict__ wlist,
                            int* __restrict__ rowmap)
{
    int t = blockIdx.x * blockDim.x + threadIdx.x;
    if (t >= T_) return;
#pragma unroll
    for (int k = 0; k < 2; ++k) {
        int e   = tope[t * 2 + k];
        int pos = atomicAdd(&counts2[e], 1);
        int r   = base[e] + pos;
        list[r]  = t;
        wlist[r] = topw[t * 2 + k];
        rowmap[t * 2 + k] = r;
    }
}

// ======================================================================
// GEMM tiling (both passes):
//   block = 256 threads (8 waves); block tile M=32, N=512
//   waves arranged 2(M) x 4(N); wave tile = 16 x 128 -> 8 WMMAs / K-step
//   B is staged to LDS ALREADY SWIZZLED into the wave32 B-fragment layout
//   so the compute phase reads whole fragments with ds_load_b128.
//   sBf[subtile][lane][j]: j<4 -> K-pair kb2+j ; j>=4 -> K-pair 8+kb2+(j-4)
//   where kb2 = (lane<16)?0:4 and column n = subtile*16 + (lane&15).
// ======================================================================

// ---------------- GEMM1: h[r,F] = relu(x[tok]·W1[e] + b1[e]), bf16 out --------
__global__ __launch_bounds__(256) void moe_gemm1(
    const float* __restrict__ x, const float* __restrict__ W1,
    const float* __restrict__ b1, const int* __restrict__ list,
    const int* __restrict__ base, const int* __restrict__ counts,
    unsigned short* __restrict__ hbuf)
{
    const int e    = blockIdx.z;
    const int cnt  = counts[e];
    const int row0 = blockIdx.y * 32;
    if (row0 >= cnt) return;
    const int rbase = base[e];
    const int n0    = blockIdx.x * 512;
    const int tid   = threadIdx.x;
    const int wave  = tid >> 5;
    const int lane  = tid & 31;
    const int wm    = wave >> 2;       // M half (0/1)
    const int wc    = wave & 3;        // N quarter (0..3)

    __shared__ unsigned int sA[32][16];       // [m][K-pair] bf16x2
    __shared__ unsigned int sBf[32][32][8];   // [subtile][lane][j] fragment layout

    v8f acc[8];
#pragma unroll
    for (int s = 0; s < 8; ++s) acc[s] = (v8f)(0.0f);

    const float* Wp    = W1 + (size_t)e * H_ * F_;
    const int    m_l   = lane & 15;
    const int    kb2   = (lane < 16) ? 0 : 4;
    const int    mbase = wm * 16;

    // A staging assignment: thread -> row am, pair base ap (2 pairs = 4 K)
    const int am = tid >> 3;
    const int ap = (tid & 7) * 2;
    int arr = row0 + am; if (arr >= cnt) arr = cnt - 1;
    const float* arow = x + (size_t)list[rbase + arr] * H_;

    for (int k0 = 0; k0 < H_; k0 += 32) {
        // ---- stage A: 32 rows x 32 K (fp32 -> bf16 pairs) ----
        {
            const float4 v = *(const float4*)(arow + k0 + 2 * ap);
            sA[am][ap]     = pk2(v.x, v.y);
            sA[am][ap + 1] = pk2(v.z, v.w);
        }
        // ---- stage B swizzled: 32K x 512N; thread loads 2 K-rows x 4 cols ----
#pragma unroll
        for (int i = 0; i < 8; ++i) {
            int c  = i * 256 + tid;        // 2048 chunks: 16 K-pairs x 128 col4
            int kp = c >> 7;               // K-pair row (0..15)
            int n4 = (c & 127) << 2;       // column 0..508
            const float* bp = Wp + (size_t)(k0 + 2 * kp) * F_ + n0 + n4;
            const float4 lo = *(const float4*)bp;
            const float4 hi = *(const float4*)(bp + F_);
            int st = n4 >> 4;
            int j  = (kp & 3) + ((kp & 8) ? 4 : 0);
            int l0 = ((kp & 4) ? 16 : 0) + (n4 & 15);
            sBf[st][l0 + 0][j] = pk2(lo.x, hi.x);
            sBf[st][l0 + 1][j] = pk2(lo.y, hi.y);
            sBf[st][l0 + 2][j] = pk2(lo.z, hi.z);
            sBf[st][l0 + 3][j] = pk2(lo.w, hi.w);
        }
        __syncthreads();

        // ---- A fragment: two contiguous 16B chunks ----
        v8ui av;
#pragma unroll
        for (int j = 0; j < 4; ++j) {
            av[j]     = sA[mbase + m_l][kb2 + j];
            av[4 + j] = sA[mbase + m_l][8 + kb2 + j];
        }
        const v16bf af = __builtin_bit_cast(v16bf, av);

        // ---- 8 WMMAs over the wave's 16x128 strip ----
#pragma unroll
        for (int s = 0; s < 8; ++s) {
            const v8ui bv = *(const v8ui*)sBf[wc * 8 + s][lane];
            const v16bf bf = __builtin_bit_cast(v16bf, bv);
            acc[s] = __builtin_amdgcn_wmma_f32_16x16x32_bf16(
                false, af, false, bf, (short)0, acc[s], false, false);
        }
        __syncthreads();
    }

    // ---- epilogue: +bias, relu, bf16 store ----
    const int mofs = (lane < 16) ? 0 : 8;
#pragma unroll
    for (int s = 0; s < 8; ++s) {
        int   nc   = n0 + wc * 128 + s * 16 + m_l;
        float bias = b1[e * F_ + nc];
#pragma unroll
        for (int v = 0; v < 8; ++v) {
            int rr = row0 + mbase + v + mofs;
            if (rr < cnt) {
                float val = fmaxf(acc[s][v] + bias, 0.f);
                hbuf[(size_t)(rbase + rr) * F_ + nc] = f2bf(val);
            }
        }
    }
}

// ---------------- GEMM2: obuf[r,H] = w[r]*(h[r]·W2[e] + b2[e]) ----------------
__global__ __launch_bounds__(256) void moe_gemm2(
    const unsigned short* __restrict__ hbuf, const float* __restrict__ W2,
    const float* __restrict__ b2, const float* __restrict__ wlist,
    const int* __restrict__ base, const int* __restrict__ counts,
    float* __restrict__ obuf)
{
    const int e    = blockIdx.z;
    const int cnt  = counts[e];
    const int row0 = blockIdx.y * 32;
    if (row0 >= cnt) return;
    const int rbase = base[e];
    const int n0    = blockIdx.x * 512;
    const int tid   = threadIdx.x;
    const int wave  = tid >> 5;
    const int lane  = tid & 31;
    const int wm    = wave >> 2;
    const int wc    = wave & 3;

    __shared__ unsigned int sA[32][16];
    __shared__ unsigned int sBf[32][32][8];

    v8f acc[8];
#pragma unroll
    for (int s = 0; s < 8; ++s) acc[s] = (v8f)(0.0f);

    const float* Wp    = W2 + (size_t)e * F_ * H_;
    const int    m_l   = lane & 15;
    const int    kb2   = (lane < 16) ? 0 : 4;
    const int    mbase = wm * 16;

    const int am = tid >> 3;
    const int ap = (tid & 7) * 2;
    int arr = row0 + am; if (arr >= cnt) arr = cnt - 1;
    const unsigned short* arow = hbuf + (size_t)(rbase + arr) * F_;

    for (int k0 = 0; k0 < F_; k0 += 32) {
        // ---- stage A (already bf16): copy 2 pairs per thread ----
        {
            const uint2 v = *(const uint2*)(arow + k0 + 2 * ap);
            *(uint2*)&sA[am][ap] = v;
        }
        // ---- stage B swizzled (fp32 -> bf16) ----
#pragma unroll
        for (int i = 0; i < 8; ++i) {
            int c  = i * 256 + tid;
            int kp = c >> 7;
            int n4 = (c & 127) << 2;
            const float* bp = Wp + (size_t)(k0 + 2 * kp) * H_ + n0 + n4;
            const float4 lo = *(const float4*)bp;
            const float4 hi = *(const float4*)(bp + H_);
            int st = n4 >> 4;
            int j  = (kp & 3) + ((kp & 8) ? 4 : 0);
            int l0 = ((kp & 4) ? 16 : 0) + (n4 & 15);
            sBf[st][l0 + 0][j] = pk2(lo.x, hi.x);
            sBf[st][l0 + 1][j] = pk2(lo.y, hi.y);
            sBf[st][l0 + 2][j] = pk2(lo.z, hi.z);
            sBf[st][l0 + 3][j] = pk2(lo.w, hi.w);
        }
        __syncthreads();

        v8ui av;
#pragma unroll
        for (int j = 0; j < 4; ++j) {
            av[j]     = sA[mbase + m_l][kb2 + j];
            av[4 + j] = sA[mbase + m_l][8 + kb2 + j];
        }
        const v16bf af = __builtin_bit_cast(v16bf, av);

#pragma unroll
        for (int s = 0; s < 8; ++s) {
            const v8ui bv = *(const v8ui*)sBf[wc * 8 + s][lane];
            const v16bf bf = __builtin_bit_cast(v16bf, bv);
            acc[s] = __builtin_amdgcn_wmma_f32_16x16x32_bf16(
                false, af, false, bf, (short)0, acc[s], false, false);
        }
        __syncthreads();
    }

    const int mofs = (lane < 16) ? 0 : 8;
#pragma unroll
    for (int s = 0; s < 8; ++s) {
        int   nc   = n0 + wc * 128 + s * 16 + m_l;
        float bias = b2[e * H_ + nc];
#pragma unroll
        for (int v = 0; v < 8; ++v) {
            int rr = row0 + mbase + v + mofs;
            if (rr < cnt) {
                float w = wlist[rbase + rr];
                obuf[(size_t)(rbase + rr) * H_ + nc] = w * (acc[s][v] + bias);
            }
        }
    }
}

// ---------------- combine: out[t] = obuf[row(t,0)] + obuf[row(t,1)] ----------------
__global__ __launch_bounds__(256) void moe_combine(
    const float* __restrict__ obuf, const int* __restrict__ rowmap,
    float* __restrict__ out)
{
    int c  = blockIdx.x * blockDim.x + threadIdx.x; // float4 chunk over [T,H]
    int t  = c >> 8;                                // H/4 = 256 chunks per token
    int n4 = c & 255;
    int r0 = rowmap[t * 2], r1 = rowmap[t * 2 + 1];
    float4 a = ((const float4*)(obuf + (size_t)r0 * H_))[n4];
    float4 b = ((const float4*)(obuf + (size_t)r1 * H_))[n4];
    float4 r; r.x = a.x + b.x; r.y = a.y + b.y; r.z = a.z + b.z; r.w = a.w + b.w;
    ((float4*)out)[c] = r;
}

extern "C" void kernel_launch(void* const* d_in, const int* in_sizes, int n_in,
                              void* d_out, int out_size, void* d_ws, size_t ws_size,
                              hipStream_t stream) {
    const float* x  = (const float*)d_in[0];
    const float* Wr = (const float*)d_in[1];
    const float* br = (const float*)d_in[2];
    const float* W1 = (const float*)d_in[3];
    const float* b1 = (const float*)d_in[4];
    const float* W2 = (const float*)d_in[5];
    const float* b2 = (const float*)d_in[6];
    float* out = (float*)d_out;

    // workspace layout (~97 MB)
    char* ws = (char*)d_ws;
    size_t off = 0;
    unsigned short* hbuf = (unsigned short*)(ws + off); off += (size_t)R_ * F_ * 2;  // 64 MB
    float* obuf  = (float*)(ws + off); off += (size_t)R_ * H_ * 4;                   // 32 MB
    int*   list  = (int*)  (ws + off); off += (size_t)R_ * 4;
    float* wlist = (float*)(ws + off); off += (size_t)R_ * 4;
    int*   rowmap= (int*)  (ws + off); off += (size_t)T_ * 2 * 4;
    int*   tope  = (int*)  (ws + off); off += (size_t)T_ * 2 * 4;
    float* topw  = (float*)(ws + off); off += (size_t)T_ * 2 * 4;
    int*   counts  = (int*)(ws + off); off += 256;
    int*   counts2 = (int*)(ws + off); off += 256;
    int*   basep   = (int*)(ws + off); off += 256;

    moe_init   <<<1, 64, 0, stream>>>(counts, counts2);
    moe_router <<<T_ / 8, 256, 0, stream>>>(x, Wr, br, tope, topw, counts);
    moe_scan   <<<1, 1, 0, stream>>>(counts, basep);
    moe_scatter<<<T_ / 256, 256, 0, stream>>>(tope, topw, basep, counts2, list, wlist, rowmap);
    moe_gemm1  <<<dim3(F_ / 512, T_ / 32, E_), 256, 0, stream>>>(x, W1, b1, list, basep, counts, hbuf);
    moe_gemm2  <<<dim3(H_ / 512, T_ / 32, E_), 256, 0, stream>>>(hbuf, W2, b2, wlist, basep, counts, obuf);
    moe_combine<<<(T_ * H_ / 4) / 256, 256, 0, stream>>>(obuf, rowmap, out);
}
